// PinConv_54202487275763
// MI455X (gfx1250) — compile-verified, matrix operand order checked by reading, not run on hardware
//
#include <hip/hip_runtime.h>
#include <hip/hip_bf16.h>

typedef __attribute__((ext_vector_type(16))) _Float16 v16h;
typedef __attribute__((ext_vector_type(8)))  _Float16 v8h;
typedef __attribute__((ext_vector_type(8)))  float    v8f;
typedef __attribute__((ext_vector_type(4)))  float    v4f;

#define DFEAT 128    // D_IN == D_H == D_OUT == 128
#define OUTLD 132    // padded f32 LDS row (breaks 2-way bank conflict at stride 128)

// ---------------------------------------------------------------------------
// Fragment gather for V_WMMA_F32_16X16X32_F16 (wave32).
// 16-bit A-matrix 16x32 layout (ISA 7.12.2): per lane, the fragment is two
// contiguous 8-half (16B) runs: [kBase..kBase+7] and [kBase+16..kBase+23],
// with kBase += 8 for lanes 16..31. B (32x16) mirrors with lanes = columns.
// Works for both global (addrspace 1) and LDS-backed generic pointers.
// ---------------------------------------------------------------------------
__device__ __forceinline__ v16h make_frag(const _Float16* rowptr, int kBase) {
  v8h lo = *(const v8h*)(rowptr + kBase);
  v8h hi = *(const v8h*)(rowptr + kBase + 16);
  v16h r;
#pragma unroll
  for (int i = 0; i < 8; ++i) { r[i] = lo[i]; r[i + 8] = hi[i]; }
  return r;
}

// ---------------------------------------------------------------------------
__global__ void zero_f32(float* p, long n) {
  long i = (long)blockIdx.x * blockDim.x + threadIdx.x;
  long stride = (long)gridDim.x * blockDim.x;
  for (; i < n; i += stride) p[i] = 0.0f;
}

__global__ void f32_to_f16(const float* s, _Float16* d, long n) {
  long i = (long)blockIdx.x * blockDim.x + threadIdx.x;
  long stride = (long)gridDim.x * blockDim.x;
  for (; i < n; i += stride) d[i] = (_Float16)s[i];
}

// ---------------------------------------------------------------------------
// GEMM1: h0 = relu(feat @ Q_w^T + Q_b)   [N,128] = [N,128] x [128,128]^T
// Block: 16 rows x 128 cols, 8 waves, wave w -> cols 16w..16w+15.
// A tile staged once in LDS; C tile transposed through padded LDS so the
// final stores are coalesced global_store_b128.
// ---------------------------------------------------------------------------
__global__ __launch_bounds__(256) void gemm1_wmma(
    const _Float16* __restrict__ featH, const _Float16* __restrict__ QwH,
    const float* __restrict__ Qb, float* __restrict__ h0, int N) {
  const int row0 = blockIdx.x * 16;
  __shared__ __align__(16) _Float16 featT[16 * DFEAT];
  __shared__ __align__(16) float    outT[16 * OUTLD];

  // stage A tile: 16 rows x 128 halfs, 16B per thread
  {
    const int i0 = threadIdx.x * 8;
    const int r = i0 >> 7, k = i0 & 127;
    const int row = row0 + r;
    v8h val = {};
    if (row < N) val = *(const v8h*)(featH + (size_t)row * DFEAT + k);
    *(v8h*)(featT + r * DFEAT + k) = val;
  }
  __syncthreads();

  const int wave = threadIdx.x >> 5;
  const int lane = threadIdx.x & 31;
  const int col0 = wave * 16;
  const int hi16 = lane >> 4;
  const int lidx = lane & 15;
  const int bcol = col0 + lidx;

  v8f c = {};
#pragma unroll
  for (int kt = 0; kt < 4; ++kt) {
    const int kBase = kt * 32 + hi16 * 8;
    v16h a = make_frag(featT + lidx * DFEAT, kBase);          // LDS
    v16h b = make_frag(QwH + (size_t)bcol * DFEAT, kBase);    // global (L2-hot)
    c = __builtin_amdgcn_wmma_f32_16x16x32_f16(
        false, a, false, b, (short)0, c, false, false);
  }

  // bias + relu, transpose through LDS
  const float bias = Qb[bcol];
#pragma unroll
  for (int r = 0; r < 8; ++r) {
    float v = c[r] + bias;
    outT[(r + hi16 * 8) * OUTLD + bcol] = v > 0.0f ? v : 0.0f;
  }
  __syncthreads();

  // coalesced store: thread t -> row t>>4, cols (t&15)*8 .. +7
  {
    const int rr = threadIdx.x >> 4;
    const int cc = (threadIdx.x & 15) * 8;
    const int row = row0 + rr;
    if (row < N) {
      const float* sp = &outT[rr * OUTLD + cc];
      v4f a0 = *(const v4f*)(sp);
      v4f a1 = *(const v4f*)(sp + 4);
      v4f* dp = (v4f*)&h0[(size_t)row * DFEAT + cc];
      dp[0] = a0; dp[1] = a1;
    }
  }
}

// ---------------------------------------------------------------------------
// Per-feature batch stats (sum, sum-of-squares over rows).
// ---------------------------------------------------------------------------
__global__ void stats128(const float* __restrict__ x, float* __restrict__ sum,
                         float* __restrict__ ssq, int N) {
  const int j = threadIdx.x & 127;
  const long stride = (long)gridDim.x * 2;
  float s = 0.0f, q = 0.0f;
  for (long row = (long)blockIdx.x * 2 + (threadIdx.x >> 7); row < N; row += stride) {
    float v = x[row * DFEAT + j];
    s += v; q += v * v;
  }
  atomicAdd(&sum[j], s);
  atomicAdd(&ssq[j], q);
}

__global__ void bn_finalize(const float* __restrict__ sum, const float* __restrict__ ssq,
                            const float* __restrict__ gamma, const float* __restrict__ beta,
                            float* __restrict__ scale, float* __restrict__ shift, float n) {
  const int j = threadIdx.x;
  const float mean = sum[j] / n;
  float var = ssq[j] / n - mean * mean;
  if (var < 0.0f) var = 0.0f;
  const float sc = rsqrtf(var + 1e-5f) * gamma[j];
  scale[j] = sc;
  shift[j] = beta[j] - mean * sc;
}

// ---------------------------------------------------------------------------
// Edge scatter with BN1 folded in: num[dst] += w_e*(h0[src]*sc+sh); den[dst]+=w_e.
// Grid-stride, 2 edges per 256-thread block per step; prefetch next h0 row
// (emits global_prefetch_b8) to hide gather latency behind the atomic stream.
// ---------------------------------------------------------------------------
__global__ __launch_bounds__(256) void edge_scatter(
    const int* __restrict__ src, const int* __restrict__ dst,
    const float* __restrict__ w, const float* __restrict__ h0,
    const float* __restrict__ scale1, const float* __restrict__ shift1,
    float* __restrict__ num, float* __restrict__ den, int E) {
  const int j = threadIdx.x & 127;
  const float sc = scale1[j];
  const float sh = shift1[j];
  const long stride = (long)gridDim.x * 2;
  for (long e = (long)blockIdx.x * 2 + (threadIdx.x >> 7); e < E; e += stride) {
    const int s = src[e];
    const int d = dst[e];
    const float we = w[e];
    const long en = e + stride;
    if (en < E) __builtin_prefetch(&h0[(size_t)src[en] * DFEAT + j], 0, 1);
    const float hv = h0[(size_t)s * DFEAT + j] * sc + sh;
    atomicAdd(&num[(size_t)d * DFEAT + j], we * hv);
    if (j == 0) atomicAdd(&den[d], we);
  }
}

// ---------------------------------------------------------------------------
// GEMM2: rst0 = relu([feat, agg] @ W_w^T + W_b), K = 256.
// agg tile built in LDS (BN1-on-the-fly fallback), feat tile staged in LDS,
// C tile transposed through padded LDS for coalesced stores.
// ---------------------------------------------------------------------------
__global__ __launch_bounds__(256) void gemm2_wmma(
    const _Float16* __restrict__ featH, const _Float16* __restrict__ WwH,
    const float* __restrict__ Wb, const float* __restrict__ num,
    const float* __restrict__ den, const float* __restrict__ h0,
    const float* __restrict__ scale1, const float* __restrict__ shift1,
    float* __restrict__ rst0, int N) {
  const int row0 = blockIdx.x * 16;
  __shared__ __align__(16) _Float16 featT[16 * DFEAT];
  __shared__ __align__(16) _Float16 aggT[16 * DFEAT];
  __shared__ __align__(16) float    outT[16 * OUTLD];

  // stage feat tile
  {
    const int i0 = threadIdx.x * 8;
    const int r = i0 >> 7, k = i0 & 127;
    const int row = row0 + r;
    v8h val = {};
    if (row < N) val = *(const v8h*)(featH + (size_t)row * DFEAT + k);
    *(v8h*)(featT + r * DFEAT + k) = val;
  }
  // build agg tile: num/den where den!=0, else BN1(h0)
  for (int idx = threadIdx.x; idx < 16 * DFEAT; idx += 256) {
    const int r = idx >> 7, j = idx & 127;
    const int row = row0 + r;
    float v = 0.0f;
    if (row < N) {
      const float d = den[row];
      v = (d != 0.0f) ? num[(size_t)row * DFEAT + j] / d
                      : h0[(size_t)row * DFEAT + j] * scale1[j] + shift1[j];
    }
    aggT[idx] = (_Float16)v;
  }
  __syncthreads();

  const int wave = threadIdx.x >> 5;
  const int lane = threadIdx.x & 31;
  const int col0 = wave * 16;
  const int hi16 = lane >> 4;
  const int lidx = lane & 15;
  const int bcol = col0 + lidx;

  v8f c = {};
#pragma unroll
  for (int kt = 0; kt < 8; ++kt) {
    const int kBase = kt * 32 + hi16 * 8;       // 16B runs never straddle k=128
    v16h a = (kBase < DFEAT)
                 ? make_frag(featT + lidx * DFEAT, kBase)
                 : make_frag(aggT + lidx * DFEAT, kBase - DFEAT);
    v16h b = make_frag(WwH + (size_t)bcol * 2 * DFEAT, kBase);
    c = __builtin_amdgcn_wmma_f32_16x16x32_f16(
        false, a, false, b, (short)0, c, false, false);
  }
  __syncthreads();   // aggT/featT dead; safe before outT reuse of LDS banks

  const float bias = Wb[bcol];
#pragma unroll
  for (int r = 0; r < 8; ++r) {
    float v = c[r] + bias;
    outT[(r + hi16 * 8) * OUTLD + bcol] = v > 0.0f ? v : 0.0f;
  }
  __syncthreads();

  {
    const int rr = threadIdx.x >> 4;
    const int cc = (threadIdx.x & 15) * 8;
    const int row = row0 + rr;
    if (row < N) {
      const float* sp = &outT[rr * OUTLD + cc];
      v4f a0 = *(const v4f*)(sp);
      v4f a1 = *(const v4f*)(sp + 4);
      v4f* dp = (v4f*)&rst0[(size_t)row * DFEAT + cc];
      dp[0] = a0; dp[1] = a1;
    }
  }
}

// ---------------------------------------------------------------------------
// y = BN2(rst0); out = y / ||y||_2 (0-norm -> 1). One row per 128-thread block.
// ---------------------------------------------------------------------------
__global__ __launch_bounds__(128) void final_norm(
    const float* __restrict__ rst0, const float* __restrict__ scale,
    const float* __restrict__ shift, float* __restrict__ out, int N) {
  const int row = blockIdx.x;
  if (row >= N) return;
  const int j = threadIdx.x;
  __shared__ float red[4];
  float y = rst0[(size_t)row * DFEAT + j] * scale[j] + shift[j];
  float sq = y * y;
#pragma unroll
  for (int off = 16; off > 0; off >>= 1) sq += __shfl_xor(sq, off, 32);
  if ((threadIdx.x & 31) == 0) red[threadIdx.x >> 5] = sq;
  __syncthreads();
  const float tot = red[0] + red[1] + red[2] + red[3];
  const float nrm = sqrtf(tot);
  out[(size_t)row * DFEAT + j] = y / (nrm == 0.0f ? 1.0f : nrm);
}

// ---------------------------------------------------------------------------
extern "C" void kernel_launch(void* const* d_in, const int* in_sizes, int n_in,
                              void* d_out, int out_size, void* d_ws, size_t ws_size,
                              hipStream_t stream) {
  const float* feat  = (const float*)d_in[0];
  const float* w     = (const float*)d_in[1];
  const float* Qw    = (const float*)d_in[2];
  const float* Qb    = (const float*)d_in[3];
  const float* Ww    = (const float*)d_in[4];
  const float* Wb    = (const float*)d_in[5];
  const float* gamma = (const float*)d_in[6];
  const float* beta  = (const float*)d_in[7];
  const int*   src   = (const int*)d_in[8];
  const int*   dst   = (const int*)d_in[9];
  float* out = (float*)d_out;

  const long N = (long)in_sizes[0] / DFEAT;
  const long E = (long)in_sizes[1];

  // Workspace layout (floats). Region [0, zeroCount) must be zero each call.
  float* ws = (float*)d_ws;
  size_t o = 0;
  float* num   = ws + o; o += (size_t)N * DFEAT;
  float* den   = ws + o; o += (size_t)N;
  float* sum1  = ws + o; o += DFEAT;
  float* ssq1  = ws + o; o += DFEAT;
  float* sum2  = ws + o; o += DFEAT;
  float* ssq2  = ws + o; o += DFEAT;
  const size_t zeroCount = o;
  float* scale1 = ws + o; o += DFEAT;
  float* shift1 = ws + o; o += DFEAT;
  float* scale2 = ws + o; o += DFEAT;
  float* shift2 = ws + o; o += DFEAT;
  float* h0    = ws + o; o += (size_t)N * DFEAT;   // pre-BN activations (BN folded downstream)
  float* rst0  = ws + o; o += (size_t)N * DFEAT;
  _Float16* featH = (_Float16*)(ws + o); o += (size_t)N * DFEAT / 2;
  _Float16* QwH   = (_Float16*)(ws + o); o += (size_t)DFEAT * DFEAT / 2;
  _Float16* WwH   = (_Float16*)(ws + o); o += (size_t)DFEAT * 2 * DFEAT / 2;

  zero_f32<<<1024, 256, 0, stream>>>(ws, (long)zeroCount);
  f32_to_f16<<<1024, 256, 0, stream>>>(feat, featH, N * DFEAT);
  f32_to_f16<<<64,   256, 0, stream>>>(Qw, QwH, (long)DFEAT * DFEAT);
  f32_to_f16<<<128,  256, 0, stream>>>(Ww, WwH, (long)DFEAT * 2 * DFEAT);

  const int tiles = (int)((N + 15) / 16);
  gemm1_wmma<<<tiles, 256, 0, stream>>>(featH, QwH, Qb, h0, (int)N);
  stats128<<<512, 256, 0, stream>>>(h0, sum1, ssq1, (int)N);
  bn_finalize<<<1, DFEAT, 0, stream>>>(sum1, ssq1, gamma, beta, scale1, shift1, (float)N);
  edge_scatter<<<65536, 256, 0, stream>>>(src, dst, w, h0, scale1, shift1, num, den, (int)E);
  gemm2_wmma<<<tiles, 256, 0, stream>>>(featH, WwH, Wb, num, den, h0, scale1, shift1,
                                        rst0, (int)N);
  stats128<<<512, 256, 0, stream>>>(rst0, sum2, ssq2, (int)N);
  bn_finalize<<<1, DFEAT, 0, stream>>>(sum2, ssq2, gamma, beta, scale2, shift2, (float)N);
  final_norm<<<(int)N, DFEAT, 0, stream>>>(rst0, scale2, shift2, out, (int)N);
}
